// HierarchicalTransformer_76201309765909
// MI455X (gfx1250) — compile-verified, matrix operand order checked by reading, not run on hardware
//
#include <hip/hip_runtime.h>
#include <hip/hip_fp16.h>

// ---------------------------------------------------------------------------
// Types for CDNA5 WMMA
// ---------------------------------------------------------------------------
typedef __attribute__((ext_vector_type(16))) _Float16 v16h;
typedef __attribute__((ext_vector_type(8)))  _Float16 v8h;
typedef __attribute__((ext_vector_type(8)))  float    v8f;
typedef __attribute__((ext_vector_type(4)))  float    f32x4;
typedef _Float16 h16_t;

#define D_MODEL 256
#define QKV_N   768
#define FF_DIM  2048
#define B_GRAPH 4
#define LEV     4
#define HPL     32
#define HL      32
#define NNODES  32768
#define NMASK   512
#define S_MCM   911            // 512 + 3*133
#define TOK_MCM (B_GRAPH * S_MCM)
#define MAXTOK  16896          // 512 sequences * 33 tokens (hop stage)

// ---------------------------------------------------------------------------
// Weight pack: W[N,K] f32  ->  fragment-contiguous f16 blocks.
// Packed layout: for k-block kb (32 K's), n-tile t (16 N's), lane l (0..31),
// 16 halfs in exactly the per-lane order V_WMMA_F32_16X16X32_F16 expects for
// its B operand (ISA 7.12.2).  A lane's whole fragment is then 32 contiguous
// bytes -> two b128 loads in the GEMM.
// ---------------------------------------------------------------------------
__global__ void wt_pack_kernel(const float* __restrict__ W, h16_t* __restrict__ Bp,
                               int N, int K) {
    const size_t gid = (size_t)blockIdx.x * blockDim.x + threadIdx.x;
    const size_t total = (size_t)K * N;
    if (gid >= total) return;
    const int e = (int)(gid & 15);          // element within fragment
    const int l = (int)((gid >> 4) & 31);   // lane
    const size_t rest = gid >> 9;
    const int tilesN = N >> 4;
    const int t  = (int)(rest % tilesN);    // n-tile
    const int kb = (int)(rest / tilesN);    // k-block (32 K's)
    const int half = l >> 4;
    const int nr = l & 15;
    const int n = t * 16 + nr;
    // element e -> K offset within the 32-wide k-block (matches A-layout pairs)
    const int kk = (e < 8) ? (half * 8 + e) : (16 + half * 8 + (e - 8));
    const int k = kb * 32 + kk;
    Bp[gid] = (h16_t)W[(size_t)n * K + k];
}

// f32 -> f16 elementwise, 4-wide (totals are multiples of 4)
__global__ void cvt_f16_kernel(const float* __restrict__ src, h16_t* __restrict__ dst,
                               size_t n4) {
    const size_t i = (size_t)blockIdx.x * blockDim.x + threadIdx.x;
    if (i >= n4) return;
    const f32x4 v = ((const f32x4*)src)[i];
    h16_t out[4] = {(h16_t)v.x, (h16_t)v.y, (h16_t)v.z, (h16_t)v.w};
    *(unsigned long long*)(dst + 4 * i) = *(const unsigned long long*)out;
}

// ---------------------------------------------------------------------------
// WMMA GEMM:  C[M,N] = A[M,K](f16, row-major) * Wpacked + bias[N]
// One wave32 per 32x32 macro-tile (2x2 register tiling, 4 f32 accumulators).
// Inner loop per 32-wide k-step: 4 b128 A loads + 4 b128 B loads feeding
// 4 hazard-free v_wmma_f32_16x16x32_f16 (A fragments reused across N-subtiles,
// B fragments reused across M-subtiles).  No predication in the loop
// (A rows clamped; out-of-range output rows are simply not stored).
// ---------------------------------------------------------------------------
__global__ void gemm_wmma_kernel(const h16_t* __restrict__ A,
                                 const h16_t* __restrict__ Bp,
                                 const float* __restrict__ bias,
                                 float* __restrict__ Cf,
                                 h16_t* __restrict__ Ch,
                                 int M, int N, int K, int relu) {
    const int lane = threadIdx.x & 31;
    const int wave = blockIdx.x * (blockDim.x >> 5) + (threadIdx.x >> 5);

    const int tilesN  = N >> 4;              // packed-B granularity (16 N's)
    const int tilesN2 = N >> 5;              // 32-wide macro tiles
    const int tilesM2 = (M + 31) >> 5;
    if (wave >= tilesM2 * tilesN2) return;   // uniform across the wave

    const int tmIdx = wave / tilesN2;
    const int tnIdx = wave % tilesN2;
    const int tm = tmIdx << 5;
    const int tn = tnIdx << 5;
    const int half = lane >> 4;              // 0: lanes 0-15, 1: lanes 16-31
    const int mr = lane & 15;                // row (A) / col (B,C) index

    int am0 = tm + mr;       if (am0 >= M) am0 = M - 1;   // clamp: dead rows
    int am1 = tm + 16 + mr;  if (am1 >= M) am1 = M - 1;
    const v8h* __restrict__ Arow0 = (const v8h*)(A + (size_t)am0 * K);
    const v8h* __restrict__ Arow1 = (const v8h*)(A + (size_t)am1 * K);

    // per-lane B fragment stream: 2 v8h per (k-block, n-tile); n-tiles
    // tnIdx*2 and tnIdx*2+1 are 64 v8h apart; k-block stride = tilesN*64.
    const v8h* __restrict__ Bfrag =
        (const v8h*)Bp + ((size_t)(tnIdx * 2) * 32 + lane) * 2;
    const size_t bstride = (size_t)tilesN * 64;  // v8h units per k-block

    v8f acc00 = {}, acc01 = {}, acc10 = {}, acc11 = {};
    union Frag { v16h v; v8h h[2]; };

#pragma unroll 2
    for (int k0 = 0; k0 < K; k0 += 32) {
        Frag a0, a1, b0, b1;
        const int ai = (k0 >> 3) + half;
        a0.h[0] = Arow0[ai];
        a0.h[1] = Arow0[ai + 2];
        a1.h[0] = Arow1[ai];
        a1.h[1] = Arow1[ai + 2];
        b0.h[0] = Bfrag[0];
        b0.h[1] = Bfrag[1];
        b1.h[0] = Bfrag[64];
        b1.h[1] = Bfrag[65];
        Bfrag += bstride;
        acc00 = __builtin_amdgcn_wmma_f32_16x16x32_f16(false, a0.v, false, b0.v,
                                                       (short)0, acc00, false, false);
        acc01 = __builtin_amdgcn_wmma_f32_16x16x32_f16(false, a0.v, false, b1.v,
                                                       (short)0, acc01, false, false);
        acc10 = __builtin_amdgcn_wmma_f32_16x16x32_f16(false, a1.v, false, b0.v,
                                                       (short)0, acc10, false, false);
        acc11 = __builtin_amdgcn_wmma_f32_16x16x32_f16(false, a1.v, false, b1.v,
                                                       (short)0, acc11, false, false);
    }

    const int n0 = tn + mr;
    const int n1 = tn + 16 + mr;
    const float bv0 = bias ? bias[n0] : 0.0f;
    const float bv1 = bias ? bias[n1] : 0.0f;
#pragma unroll
    for (int j = 0; j < 8; ++j) {
        const int m0 = tm + j + 8 * half;    // ISA C/D layout: VGPR j -> M=j (+8)
        const int m1 = m0 + 16;
        if (m0 < M) {
            float v0 = acc00[j] + bv0;
            float v1 = acc01[j] + bv1;
            if (relu) { v0 = v0 > 0.0f ? v0 : 0.0f; v1 = v1 > 0.0f ? v1 : 0.0f; }
            if (Cf) { Cf[(size_t)m0 * N + n0] = v0; Cf[(size_t)m0 * N + n1] = v1; }
            if (Ch) { Ch[(size_t)m0 * N + n0] = (h16_t)v0; Ch[(size_t)m0 * N + n1] = (h16_t)v1; }
        }
        if (m1 < M) {
            float v0 = acc10[j] + bv0;
            float v1 = acc11[j] + bv1;
            if (relu) { v0 = v0 > 0.0f ? v0 : 0.0f; v1 = v1 > 0.0f ? v1 : 0.0f; }
            if (Cf) { Cf[(size_t)m1 * N + n0] = v0; Cf[(size_t)m1 * N + n1] = v1; }
            if (Ch) { Ch[(size_t)m1 * N + n0] = (h16_t)v0; Ch[(size_t)m1 * N + n1] = (h16_t)v1; }
        }
    }
}

// ---------------------------------------------------------------------------
// Attention: per (sequence, head) block; thread-per-query two-pass softmax.
// qkv rows are [q(256) | k(256) | v(256)] per token, sequences contiguous.
// ---------------------------------------------------------------------------
template <int DH>
__global__ void attn_kernel(const float* __restrict__ qkv, float* __restrict__ o,
                            int S, int nh) {
    const int s = blockIdx.x / nh;
    const int h = blockIdx.x % nh;
    const float* base = qkv + (size_t)s * S * QKV_N;
    const float scale = rsqrtf((float)DH);

    for (int q = threadIdx.x; q < S; q += blockDim.x) {
        float qr[DH];
        const float* Qp = base + (size_t)q * QKV_N + h * DH;
#pragma unroll
        for (int j = 0; j < DH; ++j) qr[j] = Qp[j];

        float mx = -3.0e38f;
        for (int k = 0; k < S; ++k) {
            const float* Kp = base + (size_t)k * QKV_N + D_MODEL + h * DH;
            float d = 0.0f;
#pragma unroll
            for (int j = 0; j < DH; ++j) d += qr[j] * Kp[j];
            mx = fmaxf(mx, d * scale);
        }

        float acc[DH];
#pragma unroll
        for (int j = 0; j < DH; ++j) acc[j] = 0.0f;
        float sum = 0.0f;
        for (int k = 0; k < S; ++k) {
            const float* Kp = base + (size_t)k * QKV_N + D_MODEL + h * DH;
            const float* Vp = base + (size_t)k * QKV_N + 2 * D_MODEL + h * DH;
            float d = 0.0f;
#pragma unroll
            for (int j = 0; j < DH; ++j) d += qr[j] * Kp[j];
            const float w = __expf(d * scale - mx);
            sum += w;
#pragma unroll
            for (int j = 0; j < DH; ++j) acc[j] += w * Vp[j];
        }
        const float inv = 1.0f / sum;
        float* Op = o + ((size_t)s * S + q) * D_MODEL + h * DH;
#pragma unroll
        for (int j = 0; j < DH; ++j) Op[j] = acc[j] * inv;
    }
}

// ---------------------------------------------------------------------------
// Fused residual-add + LayerNorm over D=256; one wave32 per row.
// ---------------------------------------------------------------------------
__global__ void add_ln_kernel(float* __restrict__ x, const float* __restrict__ y,
                              const float* __restrict__ g, const float* __restrict__ b,
                              int rows) {
    const int wave = threadIdx.x >> 5;
    const int lane = threadIdx.x & 31;
    const int row = blockIdx.x * (blockDim.x >> 5) + wave;
    if (row >= rows) return;

    const size_t base = (size_t)row * D_MODEL;
    float v[8];
    float s = 0.0f;
#pragma unroll
    for (int k = 0; k < 8; ++k) {
        const int c = lane + 32 * k;
        v[k] = x[base + c] + y[base + c];
        s += v[k];
    }
#pragma unroll
    for (int off = 16; off > 0; off >>= 1) s += __shfl_xor(s, off, 32);
    const float mean = s * (1.0f / 256.0f);

    float vs = 0.0f;
#pragma unroll
    for (int k = 0; k < 8; ++k) { const float d = v[k] - mean; vs += d * d; }
#pragma unroll
    for (int off = 16; off > 0; off >>= 1) vs += __shfl_xor(vs, off, 32);
    const float rstd = rsqrtf(vs * (1.0f / 256.0f) + 1e-5f);

#pragma unroll
    for (int k = 0; k < 8; ++k) {
        const int c = lane + 32 * k;
        x[base + c] = (v[k] - mean) * rstd * g[c] + b[c];
    }
}

// ---------------------------------------------------------------------------
// Sequence assembly / extraction kernels
// ---------------------------------------------------------------------------
__global__ void build_hop_seq_kernel(const float* __restrict__ tokens,
                                     const int* __restrict__ hop_idx,
                                     const float* __restrict__ cls_hop,
                                     float* __restrict__ x, int m) {
    const size_t gid = (size_t)blockIdx.x * blockDim.x + threadIdx.x;
    const size_t total = (size_t)MAXTOK * D_MODEL;
    if (gid >= total) return;
    const int c = (int)(gid & 255);
    const int r = (int)(gid >> 8);
    const int t = r % 33;
    const int s = r / 33;
    float val;
    if (t == 0) {
        val = cls_hop[c];
    } else {
        const int b = s >> 7;
        const int rem = s & 127;
        const int lev = rem >> 5;
        const int hp = rem & 31;
        const int node =
            hop_idx[((((size_t)m * B_GRAPH + b) * LEV + lev) * HPL + hp) * HL + (t - 1)];
        val = tokens[((size_t)m * NNODES + node) * D_MODEL + c];
    }
    x[gid] = val;
}

__global__ void build_lseq_kernel(const float* __restrict__ xhop,
                                  const float* __restrict__ cls,
                                  float* __restrict__ dst) {
    const size_t gid = (size_t)blockIdx.x * blockDim.x + threadIdx.x;
    const size_t total = (size_t)528 * D_MODEL;
    if (gid >= total) return;
    const int c = (int)(gid & 255);
    const int r = (int)(gid >> 8);
    const int t = r % 33;
    const int s = r / 33;
    dst[gid] = (t == 0) ? cls[c]
                        : xhop[(size_t)(s * 32 + t - 1) * 33 * D_MODEL + c];
}

__global__ void build_gseq_kernel(const float* __restrict__ xlev,
                                  const float* __restrict__ cls,
                                  float* __restrict__ dst) {
    const size_t gid = (size_t)blockIdx.x * blockDim.x + threadIdx.x;
    const size_t total = (size_t)20 * D_MODEL;
    if (gid >= total) return;
    const int c = (int)(gid & 255);
    const int r = (int)(gid >> 8);
    const int t = r % 5;
    const int b = r / 5;
    dst[gid] = (t == 0) ? cls[c]
                        : xlev[(size_t)(b * 4 + t - 1) * 33 * D_MODEL + c];
}

__global__ void scatter_to_allt_kernel(const float* __restrict__ src,
                                       float* __restrict__ allt,
                                       int count, int S, int perBatch,
                                       int localBase, int mi) {
    const size_t gid = (size_t)blockIdx.x * blockDim.x + threadIdx.x;
    const size_t total = (size_t)count * D_MODEL;
    if (gid >= total) return;
    const int c = (int)(gid & 255);
    const int i = (int)(gid >> 8);
    const int b = i / perBatch;
    const int loc = i % perBatch;
    const size_t dst =
        ((size_t)b * S_MCM + NMASK + mi * 133 + localBase + loc) * D_MODEL + c;
    allt[dst] = src[(size_t)i * S * D_MODEL + c];
}

__global__ void copy_masked_kernel(const float* __restrict__ masked,
                                   float* __restrict__ allt) {
    const size_t gid = (size_t)blockIdx.x * blockDim.x + threadIdx.x;
    const size_t total = (size_t)B_GRAPH * NMASK * D_MODEL;
    if (gid >= total) return;
    const int c = (int)(gid & 255);
    const int i = (int)(gid >> 8);
    const int b = i / NMASK;
    const int j = i % NMASK;
    allt[((size_t)b * S_MCM + j) * D_MODEL + c] = masked[gid];
}

__global__ void copy_out_kernel(const float* __restrict__ allt,
                                float* __restrict__ out) {
    const size_t gid = (size_t)blockIdx.x * blockDim.x + threadIdx.x;
    const size_t total = (size_t)B_GRAPH * NMASK * D_MODEL;
    if (gid >= total) return;
    const int c = (int)(gid & 255);
    const int i = (int)(gid >> 8);
    const int b = i / NMASK;
    const int j = i % NMASK;
    out[gid] = allt[((size_t)b * S_MCM + j) * D_MODEL + c];
}

// ---------------------------------------------------------------------------
// Host-side orchestration
// ---------------------------------------------------------------------------
struct LayerW {
    const float *qkv_w, *qkv_b, *out_w, *out_b, *l1_w, *l1_b, *l2_w, *l2_b;
    const float *g1, *b1, *g2, *b2;
};

struct Scratch {
    float *qkv, *o, *y;
    h16_t *x16, *o16, *hbuf;
    h16_t *wq, *wo, *w1, *w2;
};

static inline int cdiv(long a, long b) { return (int)((a + b - 1) / b); }

static LayerW make_layer(void* const* d_in, int base, int idx) {
    LayerW w;
    w.qkv_w = (const float*)d_in[base + 0] + (size_t)idx * QKV_N * D_MODEL;
    w.qkv_b = (const float*)d_in[base + 1] + (size_t)idx * QKV_N;
    w.out_w = (const float*)d_in[base + 2] + (size_t)idx * D_MODEL * D_MODEL;
    w.out_b = (const float*)d_in[base + 3] + (size_t)idx * D_MODEL;
    w.l1_w  = (const float*)d_in[base + 4] + (size_t)idx * FF_DIM * D_MODEL;
    w.l1_b  = (const float*)d_in[base + 5] + (size_t)idx * FF_DIM;
    w.l2_w  = (const float*)d_in[base + 6] + (size_t)idx * D_MODEL * FF_DIM;
    w.l2_b  = (const float*)d_in[base + 7] + (size_t)idx * D_MODEL;
    w.g1 = (const float*)d_in[base + 8]  + (size_t)idx * D_MODEL;
    w.b1 = (const float*)d_in[base + 9]  + (size_t)idx * D_MODEL;
    w.g2 = (const float*)d_in[base + 10] + (size_t)idx * D_MODEL;
    w.b2 = (const float*)d_in[base + 11] + (size_t)idx * D_MODEL;
    return w;
}

static void run_layer(const LayerW& w, float* x, int Ntok, int S, int nseq,
                      int nh, int dh, const Scratch& sc, hipStream_t stream) {
    const int T = 256;
    // weight pack (f32 [N,K] -> fragment-contiguous f16)
    wt_pack_kernel<<<cdiv((long)QKV_N * D_MODEL, T), T, 0, stream>>>(w.qkv_w, sc.wq, QKV_N, D_MODEL);
    wt_pack_kernel<<<cdiv((long)D_MODEL * D_MODEL, T), T, 0, stream>>>(w.out_w, sc.wo, D_MODEL, D_MODEL);
    wt_pack_kernel<<<cdiv((long)FF_DIM * D_MODEL, T), T, 0, stream>>>(w.l1_w, sc.w1, FF_DIM, D_MODEL);
    wt_pack_kernel<<<cdiv((long)D_MODEL * FF_DIM, T), T, 0, stream>>>(w.l2_w, sc.w2, D_MODEL, FF_DIM);

    // QKV projection
    cvt_f16_kernel<<<cdiv((long)Ntok * D_MODEL / 4, T), T, 0, stream>>>(x, sc.x16, (size_t)Ntok * D_MODEL / 4);
    {
        int tiles = cdiv(Ntok, 32) * (QKV_N / 32);
        gemm_wmma_kernel<<<cdiv(tiles, 8), 256, 0, stream>>>(
            sc.x16, sc.wq, w.qkv_b, sc.qkv, (h16_t*)nullptr, Ntok, QKV_N, D_MODEL, 0);
    }
    // attention
    if (dh == 64) attn_kernel<64><<<nseq * nh, 128, 0, stream>>>(sc.qkv, sc.o, S, nh);
    else          attn_kernel<32><<<nseq * nh, 128, 0, stream>>>(sc.qkv, sc.o, S, nh);
    // output projection
    cvt_f16_kernel<<<cdiv((long)Ntok * D_MODEL / 4, T), T, 0, stream>>>(sc.o, sc.o16, (size_t)Ntok * D_MODEL / 4);
    {
        int tiles = cdiv(Ntok, 32) * (D_MODEL / 32);
        gemm_wmma_kernel<<<cdiv(tiles, 8), 256, 0, stream>>>(
            sc.o16, sc.wo, w.out_b, sc.y, (h16_t*)nullptr, Ntok, D_MODEL, D_MODEL, 0);
    }
    add_ln_kernel<<<cdiv(Ntok, 8), 256, 0, stream>>>(x, sc.y, w.g1, w.b1, Ntok);

    // feed-forward: l1 (+ReLU, write f16 directly) then l2
    cvt_f16_kernel<<<cdiv((long)Ntok * D_MODEL / 4, T), T, 0, stream>>>(x, sc.x16, (size_t)Ntok * D_MODEL / 4);
    {
        int tiles = cdiv(Ntok, 32) * (FF_DIM / 32);
        gemm_wmma_kernel<<<cdiv(tiles, 8), 256, 0, stream>>>(
            sc.x16, sc.w1, w.l1_b, (float*)nullptr, sc.hbuf, Ntok, FF_DIM, D_MODEL, 1);
    }
    {
        int tiles = cdiv(Ntok, 32) * (D_MODEL / 32);
        gemm_wmma_kernel<<<cdiv(tiles, 8), 256, 0, stream>>>(
            sc.hbuf, sc.w2, w.l2_b, sc.y, (h16_t*)nullptr, Ntok, D_MODEL, FF_DIM, 0);
    }
    add_ln_kernel<<<cdiv(Ntok, 8), 256, 0, stream>>>(x, sc.y, w.g2, w.b2, Ntok);
}

extern "C" void kernel_launch(void* const* d_in, const int* in_sizes, int n_in,
                              void* d_out, int out_size, void* d_ws, size_t ws_size,
                              hipStream_t stream) {
    (void)in_sizes; (void)n_in; (void)out_size; (void)ws_size;
    const float* tokens    = (const float*)d_in[0];
    const float* masked    = (const float*)d_in[1];
    const int*   hop_idx   = (const int*)d_in[2];
    const float* cls_hop   = (const float*)d_in[3];
    const float* cls_subg  = (const float*)d_in[4];
    const float* cls_graph = (const float*)d_in[5];
    // hop_p: d_in[6..17], lev_p: 18..29, graph_p: 30..41, mcm_p: 42..53

    // workspace carve-out (256B aligned)
    char* p = (char*)d_ws;
    auto alloc = [&](size_t bytes) -> void* {
        void* r = (void*)p;
        p += (bytes + 255) & ~(size_t)255;
        return r;
    };
    float* x      = (float*)alloc((size_t)MAXTOK * D_MODEL * 4);
    h16_t* x16    = (h16_t*)alloc((size_t)MAXTOK * D_MODEL * 2);
    float* qkvb   = (float*)alloc((size_t)MAXTOK * QKV_N * 4);
    float* obuf   = (float*)alloc((size_t)MAXTOK * D_MODEL * 4);
    h16_t* o16    = (h16_t*)alloc((size_t)MAXTOK * D_MODEL * 2);
    float* ybuf   = (float*)alloc((size_t)MAXTOK * D_MODEL * 4);
    h16_t* hbuf   = (h16_t*)alloc((size_t)MAXTOK * FF_DIM * 2);
    float* allt   = (float*)alloc((size_t)TOK_MCM * D_MODEL * 4);
    float* tmpseq = (float*)alloc((size_t)528 * D_MODEL * 4);
    h16_t* wq     = (h16_t*)alloc((size_t)D_MODEL * QKV_N * 2);
    h16_t* wo     = (h16_t*)alloc((size_t)D_MODEL * D_MODEL * 2);
    h16_t* w1     = (h16_t*)alloc((size_t)D_MODEL * FF_DIM * 2);
    h16_t* w2     = (h16_t*)alloc((size_t)FF_DIM * D_MODEL * 2);

    Scratch sc;
    sc.qkv = qkvb; sc.o = obuf; sc.y = ybuf;
    sc.x16 = x16; sc.o16 = o16; sc.hbuf = hbuf;
    sc.wq = wq; sc.wo = wo; sc.w1 = w1; sc.w2 = w2;

    // masked tokens -> MCM buffer
    copy_masked_kernel<<<(B_GRAPH * NMASK * D_MODEL) / 256, 256, 0, stream>>>(masked, allt);

    for (int m = 1; m <= 3; ++m) {
        const int mi = m - 1;

        // ---- hop level: 512 sequences of length 33 ----
        build_hop_seq_kernel<<<(MAXTOK * D_MODEL) / 256, 256, 0, stream>>>(
            tokens, hop_idx, cls_hop, x, m);
        for (int l = 0; l < 2; ++l)
            run_layer(make_layer(d_in, 6, m * 2 + l), x, MAXTOK, 33, 512, 4, 64, sc, stream);
        scatter_to_allt_kernel<<<512, 256, 0, stream>>>(x, allt, 512, 33, 128, 0, mi);

        // ---- subgraph level: 16 sequences of length 33 ----
        build_lseq_kernel<<<528, 256, 0, stream>>>(x, cls_subg, tmpseq);
        hipMemcpyAsync(x, tmpseq, (size_t)528 * D_MODEL * 4, hipMemcpyDeviceToDevice, stream);
        for (int l = 0; l < 2; ++l)
            run_layer(make_layer(d_in, 18, m * 2 + l), x, 528, 33, 16, 4, 64, sc, stream);
        scatter_to_allt_kernel<<<16, 256, 0, stream>>>(x, allt, 16, 33, 4, 128, mi);

        // ---- graph level: 4 sequences of length 5 ----
        build_gseq_kernel<<<20, 256, 0, stream>>>(x, cls_graph, tmpseq);
        hipMemcpyAsync(x, tmpseq, (size_t)20 * D_MODEL * 4, hipMemcpyDeviceToDevice, stream);
        for (int l = 0; l < 2; ++l)
            run_layer(make_layer(d_in, 30, m * 2 + l), x, 20, 5, 4, 4, 64, sc, stream);
        scatter_to_allt_kernel<<<4, 256, 0, stream>>>(x, allt, 4, 5, 1, 132, mi);
    }

    // ---- MCM encoder: 4 sequences of length 911, 8 heads ----
    for (int l = 0; l < 3; ++l)
        run_layer(make_layer(d_in, 42, l), allt, TOK_MCM, S_MCM, B_GRAPH, 8, 32, sc, stream);

    copy_out_kernel<<<(B_GRAPH * NMASK * D_MODEL) / 256, 256, 0, stream>>>(allt, (float*)d_out);
}
